// _Expert_23252952940690
// MI455X (gfx1250) — matrix operand order repaired and checked by reading, not compiled, not run on hardware
//
#include <hip/hip_runtime.h>

typedef __bf16 bf16_t;
typedef __attribute__((ext_vector_type(16))) __bf16        v16bf;
typedef __attribute__((ext_vector_type(8)))  __bf16        v8bf;
typedef __attribute__((ext_vector_type(8)))  float         v8f;
typedef __attribute__((ext_vector_type(4)))  unsigned int  v4u;
typedef __attribute__((ext_vector_type(4)))  unsigned int  u32x4;
typedef __attribute__((ext_vector_type(8)))  int           i32x8;
typedef __attribute__((ext_vector_type(4)))  int           i32x4;

#define NUM_EXPERTS 8
#define DMODEL      1024
#define DHIDDEN     4096
#define BM          16      // token rows per workgroup
#define NWAVES      8
#define BLOCK       (NWAVES * 32)

#define W1_ELEMS ((size_t)NUM_EXPERTS * DHIDDEN * DMODEL)   // 33.5M
#define W2_ELEMS ((size_t)NUM_EXPERTS * DMODEL * DHIDDEN)   // 33.5M

#if __has_builtin(__builtin_amdgcn_tensor_load_to_lds) && __has_builtin(__builtin_amdgcn_s_wait_tensorcnt)
#define USE_TDM 1
#else
#define USE_TDM 0
#endif

static_assert(DHIDDEN % (NWAVES * 64) == 0, "phase1 tiling");
static_assert(DMODEL  % (NWAVES * 64) == 0, "phase2 tiling");

// gelu_tanh(x) = 0.5x(1+tanh(z)) = x * sigmoid(2z),  z = c0*(x + c1*x^3)
// branch-free: v_exp_f32 (TRANS) + v_rcp_f32 + a few FMAs
__device__ __forceinline__ float gelu_tanh(float x) {
    const float km2c0 = -1.5957691216057308f;   // -2*sqrt(2/pi)
    const float c1    = 0.044715f;
    float x3 = x * x * x;
    float e  = __expf(km2c0 * (x + c1 * x3));
    return x * __builtin_amdgcn_rcpf(1.0f + e);
}

union FragU {
    v16bf v;
    struct { v4u lo; v4u hi; } u;
};

// ---------------------------------------------------------------------------
// One-pass f32 -> bf16 conversion into workspace (8 elements / thread).
// All experts' bf16 weights (128 MB) fit in MI455X's 192 MB L2, so the GEMM's
// 64x per-expert weight reuse is served from L2, not HBM.
// ---------------------------------------------------------------------------
__global__ __launch_bounds__(256)
void cvt_f32_bf16(const float* __restrict__ src, bf16_t* __restrict__ dst, long n8) {
    long i = (long)blockIdx.x * blockDim.x + threadIdx.x;
    if (i >= n8) return;
    size_t base = (size_t)i * 8;
    v8f x = *reinterpret_cast<const v8f*>(src + base);
    v8bf y;
#pragma unroll
    for (int k = 0; k < 8; ++k) y[k] = (bf16_t)x[k];
    *reinterpret_cast<v8bf*>(dst + base) = y;
}

// ---------------------------------------------------------------------------
// Fused MoE FFN: per 16-token row tile,
//   h = gelu(x @ W1[e]^T + b1[e])  (bf16, staged in LDS, 128 KB)
//   out = h @ W2[e]^T + b2[e]      (f32 to global)
// ---------------------------------------------------------------------------
__global__ __launch_bounds__(BLOCK, 1)
void moe_ffn_fused(const bf16_t* __restrict__ inp_bf,
                   const bf16_t* __restrict__ w1bf,
                   const float*  __restrict__ b1,
                   const bf16_t* __restrict__ w2bf,
                   const float*  __restrict__ b2,
                   const int*    __restrict__ counts,
                   float*        __restrict__ out,
                   int           n_tokens)
{
    extern __shared__ char smem_raw[];
    bf16_t* sh_h = reinterpret_cast<bf16_t*>(smem_raw);                     // [BM][DHIDDEN] (128 KB)
    bf16_t* sh_a = reinterpret_cast<bf16_t*>(smem_raw + BM * DHIDDEN * 2);  // [BM][DMODEL]  (32 KB)

    const int tid  = threadIdx.x;
    const int lane = tid & 31;
    const int wave = tid >> 5;
    const int row0 = blockIdx.x * BM;

    // ---- expert routing: eid = #(cumsum(counts) <= row0) ----
    int eid = 0;
    {
        int cum = 0;
#pragma unroll
        for (int e = 0; e < NUM_EXPERTS; ++e) {
            cum += counts[e];
            eid += (row0 >= cum) ? 1 : 0;
        }
    }
    if (eid >= NUM_EXPERTS) {
        for (int i = tid; i < BM * DMODEL; i += BLOCK)
            out[(size_t)row0 * DMODEL + i] = 0.0f;
        return;
    }

    const bf16_t* W1e = w1bf + (size_t)eid * DHIDDEN * DMODEL;
    const float*  B1e = b1   + (size_t)eid * DHIDDEN;
    const bf16_t* W2e = w2bf + (size_t)eid * DMODEL * DHIDDEN;
    const float*  B2e = b2   + (size_t)eid * DMODEL;

    // ---- phase 0: stage input tile (16x1024 bf16, 32 KB) into LDS ----
#if USE_TDM
    if (wave == 0) {
        // Tensor DMA descriptor (ISA ch.8): 2-D, data_size=2B,
        // tile = [1024 x 16] of tensor [1024 x n_tokens], row stride 1024.
        unsigned long long ga =
            (unsigned long long)(const void*)(inp_bf + (size_t)row0 * DMODEL);
        u32x4 g0;
        g0[0] = 1u;                                  // count=1 (valid user descriptor)
        g0[1] = (unsigned)(BM * DHIDDEN * 2);        // lds_addr = sh_a byte offset
        g0[2] = (unsigned)(ga & 0xFFFFFFFFu);        // global_addr[31:0]
        g0[3] = (unsigned)((ga >> 32) & 0x1FFFFFFu)  // global_addr[56:32]
              | (2u << 30);                          // type = 2 ("image")
        i32x8 g1;
        g1[0] = 0x00010000;                          // data_size=1 (2 bytes/elem)
        g1[1] = (int)((DMODEL & 0xFFFF) << 16);      // tensor_dim0 lo16 (1024)
        g1[2] = (int)(((unsigned)DMODEL >> 16) & 0xFFFF)
              | (int)((n_tokens & 0xFFFF) << 16);    // tensor_dim0 hi16 | tensor_dim1 lo16
        g1[3] = (int)(((unsigned)n_tokens >> 16) & 0xFFFF)
              | (int)(DMODEL << 16);                 // tensor_dim1 hi16 | tile_dim0 (1024)
        g1[4] = BM;                                  // tile_dim1=16, tile_dim2=0
        g1[5] = DMODEL;                              // tensor_dim0_stride lo32 (1024)
        g1[6] = 0;                                   // stride hi16 | dim1_stride lo16
        g1[7] = 0;
        i32x4 g2 = {0, 0, 0, 0};
        i32x4 g3 = {0, 0, 0, 0};
        i32x8 g4 = {0, 0, 0, 0, 0, 0, 0, 0};        // extra group (6-arg toolchain form)
        __builtin_amdgcn_tensor_load_to_lds(g0, g1, g2, g3, g4, 0);
        __builtin_amdgcn_s_wait_tensorcnt(0);
    }
#else
    for (int i = tid; i < (BM * DMODEL) / 8; i += BLOCK) {
        *reinterpret_cast<v4u*>(sh_a + (size_t)i * 8) =
            *reinterpret_cast<const v4u*>(inp_bf + (size_t)row0 * DMODEL + (size_t)i * 8);
    }
#endif
    __syncthreads();

    // WMMA 16x16x32 bf16 lane mapping used here (A/B in contiguous k order; C per ISA 7.12.2):
    //   A frag: lane<16 -> row lane,    K {k0..k0+15}
    //           lane>=16 -> row lane-16, K {k0+16..k0+31}
    //   B frag: lane<16 -> col lane, K {k0..k0+15}; lane>=16 -> col lane-16, K {k0+16..k0+31}
    //   C tile: VGPR r, lane l -> (M = r + (l>=16 ? 8 : 0), N = l%16)
    const int halfA = (lane >= 16) ? 8  : 0;
    const int halfB = (lane >= 16) ? 16 : 0;
    const int nlane = lane & 15;

    // ---- phase 1: h = gelu(x @ W1^T + b1) -> sh_h (bf16) ----
    for (int g = 0; g < 8; ++g) {
        const int nbase = wave * 512 + g * 64;
        __builtin_prefetch(W1e + (size_t)nbase * DMODEL, 0, 1);   // global_prefetch_b8
        v8f acc[4] = {};
        for (int k0 = 0; k0 < DMODEL; k0 += 32) {
            FragU af;
            const bf16_t* ar = sh_a + nlane * DMODEL + k0;
            af.u.lo = *reinterpret_cast<const v4u*>(ar + halfB);        // ds_load_b128
            af.u.hi = *reinterpret_cast<const v4u*>(ar + halfB + 8);    // ds_load_b128
#pragma unroll
            for (int j = 0; j < 4; ++j) {
                const bf16_t* wr = W1e + (size_t)(nbase + j * 16 + nlane) * DMODEL + k0 + halfB;
                FragU bfr;
                const v4u* wp = reinterpret_cast<const v4u*>(wr);
                bfr.u.lo = wp[0];                                       // global_load_b128
                bfr.u.hi = wp[1];                                       // global_load_b128
                acc[j] = __builtin_amdgcn_wmma_f32_16x16x32_bf16(
                    false, af.v, false, bfr.v, (short)0, acc[j], false, false);
            }
        }
#pragma unroll
        for (int j = 0; j < 4; ++j) {
            const int n = nbase + j * 16 + nlane;
            const float bias = B1e[n];
#pragma unroll
            for (int r = 0; r < 8; ++r) {
                float x = acc[j][r] + bias;
                sh_h[(halfA + r) * DHIDDEN + n] = (bf16_t)gelu_tanh(x);
            }
        }
    }
    __syncthreads();

    // ---- phase 2: out = h @ W2^T + b2 -> global (f32) ----
    for (int g = 0; g < 2; ++g) {
        const int nbase = wave * 128 + g * 64;
        __builtin_prefetch(W2e + (size_t)nbase * DHIDDEN, 0, 1);
        v8f acc[4] = {};
        for (int k0 = 0; k0 < DHIDDEN; k0 += 32) {
            FragU af;
            const bf16_t* ar = sh_h + nlane * DHIDDEN + k0;
            af.u.lo = *reinterpret_cast<const v4u*>(ar + halfB);
            af.u.hi = *reinterpret_cast<const v4u*>(ar + halfB + 8);
#pragma unroll
            for (int j = 0; j < 4; ++j) {
                const bf16_t* wr = W2e + (size_t)(nbase + j * 16 + nlane) * DHIDDEN + k0 + halfB;
                FragU bfr;
                const v4u* wp = reinterpret_cast<const v4u*>(wr);
                bfr.u.lo = wp[0];
                bfr.u.hi = wp[1];
                acc[j] = __builtin_amdgcn_wmma_f32_16x16x32_bf16(
                    false, af.v, false, bfr.v, (short)0, acc[j], false, false);
            }
        }
#pragma unroll
        for (int j = 0; j < 4; ++j) {
            const int n = nbase + j * 16 + nlane;
            const float bias = B2e[n];
#pragma unroll
            for (int r = 0; r < 8; ++r) {
                out[(size_t)(row0 + halfA + r) * DMODEL + n] = acc[j][r] + bias;
            }
        }
    }
}

extern "C" void kernel_launch(void* const* d_in, const int* in_sizes, int n_in,
                              void* d_out, int out_size, void* d_ws, size_t ws_size,
                              hipStream_t stream) {
    (void)n_in; (void)out_size; (void)ws_size;   // needs ws_size >= ~151 MB

    const float* inp    = (const float*)d_in[0];
    const float* w1     = (const float*)d_in[1];
    const float* b1     = (const float*)d_in[2];
    const float* w2     = (const float*)d_in[3];
    const float* b2     = (const float*)d_in[4];
    const int*   counts = (const int*)d_in[5];
    float*       out    = (float*)d_out;

    const int T = in_sizes[0] / DMODEL;          // 8192

    bf16_t* w1bf   = (bf16_t*)d_ws;
    bf16_t* w2bf   = w1bf + W1_ELEMS;
    bf16_t* inp_bf = w2bf + W2_ELEMS;

    // 1) pre-convert weights + input to bf16 (stream-ordered before the GEMM)
    {
        long n8_1 = (long)(W1_ELEMS / 8);
        long n8_2 = (long)(W2_ELEMS / 8);
        long n8_i = (long)((size_t)T * DMODEL / 8);
        int  blk  = 256;
        cvt_f32_bf16<<<(int)((n8_1 + blk - 1) / blk), blk, 0, stream>>>(w1, w1bf, n8_1);
        cvt_f32_bf16<<<(int)((n8_2 + blk - 1) / blk), blk, 0, stream>>>(w2, w2bf, n8_2);
        cvt_f32_bf16<<<(int)((n8_i + blk - 1) / blk), blk, 0, stream>>>(inp, inp_bf, n8_i);
    }

    // 2) fused grouped-GEMM + GELU + grouped-GEMM
    const int grid = T / BM;                     // 512 workgroups

    const size_t shmem = (size_t)BM * DHIDDEN * sizeof(bf16_t)
                       + (size_t)BM * DMODEL  * sizeof(bf16_t); // 160 KB
    (void)hipFuncSetAttribute((const void*)moe_ffn_fused,
                              hipFuncAttributeMaxDynamicSharedMemorySize, (int)shmem);

    moe_ffn_fused<<<grid, BLOCK, shmem, stream>>>(inp_bf, w1bf, b1, w2bf, b2, counts, out, T);
}
